// MotionTransformerDecoder_55473797595856
// MI455X (gfx1250) — matrix operand order repaired and checked by reading, not compiled, not run on hardware
//
#include <hip/hip_runtime.h>
#include <cmath>

// ---------------------------------------------------------------------------
// MI455X (gfx1250): bf16 WMMA GEMMs (v_wmma_f32_16x16x32_bf16) for all
// projections/FFN/heads; A-tiles staged via CDNA5 async global->LDS DMA
// (global_load_async_to_lds_b128 + s_wait_asynccnt); VALU softmax attention
// with LDS-staged K/V. wave32 layouts per CDNA5 ISA 7.12.2.
// ---------------------------------------------------------------------------

typedef __bf16 bf16;
typedef __attribute__((ext_vector_type(16))) bf16  v16bf;
typedef __attribute__((ext_vector_type(8)))  bf16  v8bf;
typedef __attribute__((ext_vector_type(2)))  bf16  bf16x2;
typedef __attribute__((ext_vector_type(8)))  float v8f;

#define B_   128
#define S_   200
#define D_   512
#define H_   8
#define DK_  64
#define DFF_ 2048

__device__ __forceinline__ float gelu_f(float x) {
  // exact (erf) GELU, matches torch nn.GELU default
  return 0.5f * x * (1.0f + erff(x * 0.7071067811865476f));
}

__device__ __forceinline__ uint32_t lds_addr_of(const void* p) {
  // generic LDS pointer = {SHARED_BASE, lds_offset[31:0]}; AS3 value = low 32b
  return (uint32_t)(uintptr_t)p;
}

// ---------------------------------------------------------------------------
// Generic WMMA GEMM:  C[M,N] = epi( A[M,K](bf16) @ Bw[K,N](f32) + bias[N] )
// BM=BN=128, BK=32; 256 threads = 8 waves (2x4); each wave owns 64x32 =
// 4x2 C-tiles, one v_wmma_f32_16x16x32_bf16 per tile per K-step.
// A tile: async DMA global->LDS (ASYNCcnt). B tile: regs, f32->bf16 packed
// v_cvt_pk_bf16_f32 + ds_store_b32. Double-buffered LDS.
// Requires M % 128 == 0, K % 32 == 0 (true for all uses). N arbitrary.
// ---------------------------------------------------------------------------
template<bool GELU, bool ADD_PE, bool STORE_F32, bool STORE_BF16>
__global__ __launch_bounds__(256)
void k_gemm(const bf16* __restrict__ A, const float* __restrict__ Bw,
            const float* __restrict__ bias, float* __restrict__ Cf,
            bf16* __restrict__ Cbf, const float* __restrict__ pe,
            int M, int N, int K)
{
  __shared__ __align__(16) bf16 sA[2][128][40];   // [m][k], 80B padded rows
  __shared__ __align__(16) bf16 sB[2][128][40];   // [n][k] (transposed stage)

  const int tid  = threadIdx.x;
  const int lane = tid & 31;
  const int wav  = tid >> 5;
  const int wm   = wav >> 2;          // 0..1 (M direction)
  const int wn   = wav & 3;           // 0..3 (N direction)
  const int m0   = blockIdx.y * 128;
  const int n0   = blockIdx.x * 128;

  // A staging: 128 rows x 32 cols, 32B per thread via async DMA
  const int arow = tid >> 1, acol = (tid & 1) * 16;
  // B staging: each thread owns 2 adjacent k rows x 8 n cols
  const int bk2 = (tid >> 4) * 2;     // 0,2,..,30
  const int bng = (tid & 15) * 8;     // 0,8,..,120

  const v8f vz = {0.f, 0.f, 0.f, 0.f, 0.f, 0.f, 0.f, 0.f};
  v8f acc[4][2];
#pragma unroll
  for (int i = 0; i < 4; ++i)
#pragma unroll
    for (int j = 0; j < 2; ++j) acc[i][j] = vz;

  union V16 { v16bf v; v8bf h[2]; };

  const uint32_t ldsA[2] = { lds_addr_of(&sA[0][arow][acol]),
                             lds_addr_of(&sA[1][arow][acol]) };
  const bf16* aRow = A + (size_t)(m0 + arow) * K + acol;

  // CDNA5 async DMA: copies 32B (global->LDS); offset applies to both sides.
  auto asyncA = [&](int kt, int buf) {
    const uint64_t ga = (uint64_t)(uintptr_t)(aRow + kt * 32);
    asm volatile(
        "global_load_async_to_lds_b128 %0, %1, off\n\t"
        "global_load_async_to_lds_b128 %0, %1, off offset:16"
        :: "v"(ldsA[buf]), "v"(ga) : "memory");
  };
  auto waitAsync = [&]() {
    asm volatile("s_wait_asynccnt 0x0" ::: "memory");
  };

  float bP0[8], bP1[8];
  auto fetchB = [&](int kt) {
    const float* r0 = Bw + (size_t)(kt * 32 + bk2) * N + n0 + bng;
    const float* r1 = r0 + N;
    if (((N & 3) == 0) && (n0 + bng + 8 <= N)) {
#pragma unroll
      for (int j = 0; j < 2; ++j) {
        float4 f0 = *(const float4*)(r0 + 4 * j);
        float4 f1 = *(const float4*)(r1 + 4 * j);
        bP0[4*j+0] = f0.x; bP0[4*j+1] = f0.y; bP0[4*j+2] = f0.z; bP0[4*j+3] = f0.w;
        bP1[4*j+0] = f1.x; bP1[4*j+1] = f1.y; bP1[4*j+2] = f1.z; bP1[4*j+3] = f1.w;
      }
    } else {
#pragma unroll
      for (int j = 0; j < 8; ++j) {
        const int n = n0 + bng + j;
        bP0[j] = (n < N) ? r0[j] : 0.0f;
        bP1[j] = (n < N) ? r1[j] : 0.0f;
      }
    }
  };
  auto stageB = [&](int buf) {
#pragma unroll
    for (int j = 0; j < 8; ++j) {        // pack (k,k+1) -> v_cvt_pk_bf16_f32
      bf16x2 p;
      p.x = (bf16)bP0[j];
      p.y = (bf16)bP1[j];
      *(bf16x2*)&sB[buf][bng + j][bk2] = p;
    }
  };

  asyncA(0, 0);
  fetchB(0); stageB(0);
  waitAsync();
  __syncthreads();

  const int nk   = K >> 5;
  const int hlf  = lane >> 4;            // lane-half selects K sub-range
  const int mrow = wm * 64 + (lane & 15);
  const int ncol = wn * 32 + (lane & 15);

  for (int kt = 0; kt < nk; ++kt) {
    const int cur = kt & 1;
    if (kt + 1 < nk) {
      asyncA(kt + 1, cur ^ 1);           // DMA overlaps this step's WMMAs
      if (kt + 2 < nk)                   // L2 prefetch of next weight tile
        __builtin_prefetch(Bw + (size_t)((kt + 2) * 32 + bk2) * N + n0 + bng, 0, 1);
      fetchB(kt + 1);
    }

    // A 16x32 bf16 fragment: lane M=lane%16; K = h*16 + e
    V16 af[4], bfr[2];
#pragma unroll
    for (int mt = 0; mt < 4; ++mt) {
      af[mt].h[0] = *(const v8bf*)&sA[cur][mrow + mt * 16][hlf * 16];
      af[mt].h[1] = *(const v8bf*)&sA[cur][mrow + mt * 16][hlf * 16 + 8];
    }
    // B 32x16 bf16 fragment: lane N=lane%16; K = h*16 + e
#pragma unroll
    for (int nt = 0; nt < 2; ++nt) {
      bfr[nt].h[0] = *(const v8bf*)&sB[cur][ncol + nt * 16][hlf * 16];
      bfr[nt].h[1] = *(const v8bf*)&sB[cur][ncol + nt * 16][hlf * 16 + 8];
    }
#pragma unroll
    for (int mt = 0; mt < 4; ++mt)
#pragma unroll
      for (int nt = 0; nt < 2; ++nt)
        acc[mt][nt] = __builtin_amdgcn_wmma_f32_16x16x32_bf16(
            false, af[mt].v, false, bfr[nt].v, (short)0, acc[mt][nt],
            false, false);

    if (kt + 1 < nk) {
      stageB(cur ^ 1);
      waitAsync();                       // next A tile resident before barrier
    }
    __syncthreads();
  }

  // Epilogue. C/D layout: VGPR r -> M = (lane/16)*8 + r, N = lane%16.
#pragma unroll
  for (int nt = 0; nt < 2; ++nt) {
    const int n = n0 + wn * 32 + nt * 16 + (lane & 15);
    if (n >= N) continue;
    const float bv = bias[n];
    const float pv = ADD_PE ? pe[n] : 0.0f;
#pragma unroll
    for (int mt = 0; mt < 4; ++mt) {
#pragma unroll
      for (int r = 0; r < 8; ++r) {
        const int m = m0 + wm * 64 + mt * 16 + hlf * 8 + r;
        float val = acc[mt][nt][r] + bv;
        if (GELU)   val = gelu_f(val);
        if (ADD_PE) val += pv;
        const size_t idx = (size_t)m * N + n;
        if (STORE_F32)  Cf[idx]  = val;
        if (STORE_BF16) Cbf[idx] = (bf16)val;
      }
    }
  }
}

// ---------------------------------------------------------------------------
// Attention: one block per (b,h). K/V staged in LDS (bf16), f32 softmax with
// temporal bias, context written as bf16 for the WMMA O-projection.
// ---------------------------------------------------------------------------
__global__ __launch_bounds__(256)
void k_attn(const bf16* __restrict__ q, const bf16* __restrict__ k,
            const bf16* __restrict__ v, const float* __restrict__ bias,
            bf16* __restrict__ ctx)
{
  __shared__ bf16  sK[S_][DK_];
  __shared__ bf16  sV[S_][DK_];
  __shared__ float sQ[DK_];
  __shared__ float sc[256];
  __shared__ float red[256];
  __shared__ float sCtx[4][DK_];

  const int bh = blockIdx.x;
  const int b = bh >> 3, h = bh & 7;
  const int t = threadIdx.x;
  const size_t rowbase = (size_t)b * S_;

  for (int idx = t; idx < S_ * DK_; idx += 256) {
    const int j = idx >> 6, d = idx & 63;
    const size_t g = (rowbase + j) * D_ + h * DK_ + d;
    sK[j][d] = k[g];
    sV[j][d] = v[g];
  }
  __syncthreads();

  const float* bp = bias + (size_t)h * S_ * S_;
  const int d = t & 63, c = t >> 6;

  for (int i = 0; i < S_; ++i) {
    if (t < DK_) sQ[t] = (float)q[(rowbase + i) * D_ + h * DK_ + t];
    __syncthreads();

    float sv = -3.0e38f;
    if (t < S_) {
      float dot = 0.f;
#pragma unroll
      for (int dd = 0; dd < DK_; ++dd) dot += sQ[dd] * (float)sK[t][dd];
      sv = dot * 0.125f + bp[i * S_ + t];          // scale = 1/sqrt(64)
    }
    red[t] = sv;
    __syncthreads();
    for (int off = 128; off > 0; off >>= 1) {
      if (t < off) red[t] = fmaxf(red[t], red[t + off]);
      __syncthreads();
    }
    const float mx = red[0];
    __syncthreads();

    const float e = (t < S_) ? expf(sv - mx) : 0.f;
    sc[t] = e; red[t] = e;
    __syncthreads();
    for (int off = 128; off > 0; off >>= 1) {
      if (t < off) red[t] += red[t + off];
      __syncthreads();
    }
    const float inv = 1.0f / red[0];
    __syncthreads();

    float p = 0.f;
    for (int jj = 0; jj < 50; ++jj) {              // 4 chunks of 50 rows
      const int j = c * 50 + jj;
      p += sc[j] * (float)sV[j][d];
    }
    sCtx[c][d] = p;
    __syncthreads();
    if (c == 0) {
      const float r = (sCtx[0][d] + sCtx[1][d] + sCtx[2][d] + sCtx[3][d]) * inv;
      ctx[(rowbase + i) * D_ + h * DK_ + d] = (bf16)r;
    }
    __syncthreads();
  }
}

// ---------------------------------------------------------------------------
// Residual + LayerNorm (eps 1e-5); writes f32 residual stream and bf16 copy.
// One block per row of 512. In-place safe for base==xf (row-private).
// ---------------------------------------------------------------------------
__global__ __launch_bounds__(256)
void k_ln(const float* base, const float* __restrict__ delta,
          const float* __restrict__ g, const float* __restrict__ be,
          float* xf, bf16* __restrict__ xbf)
{
  __shared__ float r1[256], r2[256];
  const size_t row = blockIdx.x;
  const int t = threadIdx.x;
  const float* pb = base  + row * D_;
  const float* pd = delta + row * D_;
  const float v0 = pb[t] + pd[t];
  const float v1 = pb[t + 256] + pd[t + 256];
  r1[t] = v0 + v1;
  r2[t] = v0 * v0 + v1 * v1;
  __syncthreads();
  for (int off = 128; off > 0; off >>= 1) {
    if (t < off) { r1[t] += r1[t + off]; r2[t] += r2[t + off]; }
    __syncthreads();
  }
  const float mean = r1[0] * (1.0f / D_);
  const float var  = r2[0] * (1.0f / D_) - mean * mean;
  const float rstd = rsqrtf(var + 1e-5f);
  const float o0 = (v0 - mean) * rstd * g[t]       + be[t];
  const float o1 = (v1 - mean) * rstd * g[t + 256] + be[t + 256];
  xf[row * D_ + t]        = o0;
  xf[row * D_ + t + 256]  = o1;
  xbf[row * D_ + t]       = (bf16)o0;
  xbf[row * D_ + t + 256] = (bf16)o1;
}

// Sinusoidal positional encoding table [S, D]
__global__ void k_pe(float* __restrict__ pe) {
  const int idx = blockIdx.x * 256 + threadIdx.x;
  if (idx >= S_ * (D_ / 2)) return;
  const int s = idx / (D_ / 2), t2 = idx % (D_ / 2);
  const float ang = (float)s * expf((float)(2 * t2) * (-0.017988946039016f)); // -ln(1e4)/512
  pe[s * D_ + 2 * t2]     = sinf(ang);
  pe[s * D_ + 2 * t2 + 1] = cosf(ang);
}

// z = concat(z_style, z_skill) as bf16 [128, 256]
__global__ void k_zcat(const float* __restrict__ zs, const float* __restrict__ zk,
                       bf16* __restrict__ zbf) {
  const int idx = blockIdx.x * 256 + threadIdx.x;   // 128*256
  const int b = idx >> 8, c = idx & 255;
  const float v = (c < 128) ? zs[b * 128 + c] : zk[b * 128 + (c - 128)];
  zbf[idx] = (bf16)v;
}

// ---------------------------------------------------------------------------
extern "C" void kernel_launch(void* const* d_in, const int* in_sizes, int n_in,
                              void* d_out, int out_size, void* d_ws, size_t ws_size,
                              hipStream_t stream)
{
  (void)in_sizes; (void)n_in; (void)out_size; (void)ws_size;
  const float* z_style   = (const float*)d_in[0];
  const float* z_skill   = (const float*)d_in[1];
  const float* W_lat     = (const float*)d_in[2];
  const float* b_lat     = (const float*)d_in[3];
  const float* temp_bias = (const float*)d_in[4];
  const float* Wq = (const float*)d_in[5];  const float* bq = (const float*)d_in[6];
  const float* Wk = (const float*)d_in[7];  const float* bk = (const float*)d_in[8];
  const float* Wv = (const float*)d_in[9];  const float* bv = (const float*)d_in[10];
  const float* Wo = (const float*)d_in[11]; const float* bo = (const float*)d_in[12];
  const float* g1 = (const float*)d_in[13]; const float* be1 = (const float*)d_in[14];
  const float* g2 = (const float*)d_in[15]; const float* be2 = (const float*)d_in[16];
  const float* W1 = (const float*)d_in[17]; const float* bf1 = (const float*)d_in[18];
  const float* W2 = (const float*)d_in[19]; const float* bf2 = (const float*)d_in[20];
  const float* Wp1 = (const float*)d_in[21]; const float* bp1 = (const float*)d_in[22];
  const float* Wp2 = (const float*)d_in[23]; const float* bp2 = (const float*)d_in[24];
  const float* Ws1 = (const float*)d_in[25]; const float* bs1 = (const float*)d_in[26];
  const float* Ws2 = (const float*)d_in[27]; const float* bs2 = (const float*)d_in[28];

  const int M = B_ * S_;                       // 25600 rows
  char* w = (char*)d_ws;
  auto alloc = [&](size_t bytes) -> char* {
    char* p = w; w += (bytes + 255) & ~(size_t)255; return p;
  };
  bf16*  zbf  = (bf16*) alloc((size_t)128 * 256 * sizeof(bf16));
  float* pe   = (float*)alloc((size_t)S_ * D_ * sizeof(float));
  float* xf   = (float*)alloc((size_t)M * D_ * sizeof(float));
  bf16*  xbf  = (bf16*) alloc((size_t)M * D_ * sizeof(bf16));
  bf16*  qbf  = (bf16*) alloc((size_t)M * D_ * sizeof(bf16));
  bf16*  kbf  = (bf16*) alloc((size_t)M * D_ * sizeof(bf16));
  bf16*  vbf  = (bf16*) alloc((size_t)M * D_ * sizeof(bf16));
  bf16*  cbf  = (bf16*) alloc((size_t)M * D_ * sizeof(bf16));
  float* tmp  = (float*)alloc((size_t)M * D_ * sizeof(float));
  bf16*  h1bf = (bf16*) alloc((size_t)M * DFF_ * sizeof(bf16));
  bf16*  t1bf = (bf16*) alloc((size_t)M * 256 * sizeof(bf16));
  bf16*  s1bf = (bf16*) alloc((size_t)128 * 256 * sizeof(bf16));

  k_zcat<<<128, 256, 0, stream>>>(z_style, z_skill, zbf);
  k_pe<<<(S_ * (D_ / 2) + 255) / 256, 256, 0, stream>>>(pe);

  // x = gelu(z @ W_lat + b_lat) + pe  -- reshape is flat-identity: dst = m*(S*D)+n
  k_gemm<true, true, true, true><<<dim3(102400 / 128, 1), 256, 0, stream>>>(
      zbf, W_lat, b_lat, xf, xbf, pe, 128, 102400, 256);

  const dim3 gP(D_ / 128, M / 128);            // (4, 200) for D-wide GEMMs
  for (int l = 0; l < 6; ++l) {
    const float* Wq_l = Wq + (size_t)l * D_ * D_;   const float* bq_l = bq + (size_t)l * D_;
    const float* Wk_l = Wk + (size_t)l * D_ * D_;   const float* bk_l = bk + (size_t)l * D_;
    const float* Wv_l = Wv + (size_t)l * D_ * D_;   const float* bv_l = bv + (size_t)l * D_;
    const float* Wo_l = Wo + (size_t)l * D_ * D_;   const float* bo_l = bo + (size_t)l * D_;
    const float* W1_l = W1 + (size_t)l * D_ * DFF_; const float* b1_l = bf1 + (size_t)l * DFF_;
    const float* W2_l = W2 + (size_t)l * DFF_ * D_; const float* b2_l = bf2 + (size_t)l * D_;
    const float* g1_l = g1 + (size_t)l * D_; const float* be1_l = be1 + (size_t)l * D_;
    const float* g2_l = g2 + (size_t)l * D_; const float* be2_l = be2 + (size_t)l * D_;

    k_gemm<false, false, false, true><<<gP, 256, 0, stream>>>(
        xbf, Wq_l, bq_l, nullptr, qbf, nullptr, M, D_, D_);
    k_gemm<false, false, false, true><<<gP, 256, 0, stream>>>(
        xbf, Wk_l, bk_l, nullptr, kbf, nullptr, M, D_, D_);
    k_gemm<false, false, false, true><<<gP, 256, 0, stream>>>(
        xbf, Wv_l, bv_l, nullptr, vbf, nullptr, M, D_, D_);

    k_attn<<<B_ * H_, 256, 0, stream>>>(qbf, kbf, vbf, temp_bias, cbf);

    k_gemm<false, false, true, false><<<gP, 256, 0, stream>>>(
        cbf, Wo_l, bo_l, tmp, nullptr, nullptr, M, D_, D_);
    k_ln<<<M, 256, 0, stream>>>(xf, tmp, g1_l, be1_l, xf, xbf);

    k_gemm<true, false, false, true><<<dim3(DFF_ / 128, M / 128), 256, 0, stream>>>(
        xbf, W1_l, b1_l, nullptr, h1bf, nullptr, M, DFF_, D_);
    k_gemm<false, false, true, false><<<gP, 256, 0, stream>>>(
        h1bf, W2_l, b2_l, tmp, nullptr, nullptr, M, D_, DFF_);
    k_ln<<<M, 256, 0, stream>>>(xf, tmp, g2_l, be2_l, xf, xbf);
  }

  // trajectory head -> d_out[0 : 25600*6)
  k_gemm<true, false, false, true><<<dim3(2, M / 128), 256, 0, stream>>>(
      xbf, Wp1, bp1, nullptr, t1bf, nullptr, M, 256, D_);
  k_gemm<false, false, true, false><<<dim3(1, M / 128), 256, 0, stream>>>(
      t1bf, Wp2, bp2, (float*)d_out, nullptr, nullptr, M, 6, 256);

  // signature head -> d_out[25600*6 : +128*5)
  k_gemm<true, false, false, true><<<dim3(2, 1), 256, 0, stream>>>(
      zbf, Ws1, bs1, nullptr, s1bf, nullptr, 128, 256, 256);
  k_gemm<false, false, true, false><<<dim3(1, 1), 256, 0, stream>>>(
      s1bf, Ws2, bs2, (float*)d_out + (size_t)M * 6, nullptr, nullptr, 128, 5, 256);
}
